// SinkhornChannelMixer_25950192403055
// MI455X (gfx1250) — matrix operand order repaired and checked by reading, not compiled, not run on hardware
//
#include <hip/hip_runtime.h>
#include <hip/hip_bf16.h>
#include <math.h>

// ---------------------------------------------------------------------------
// SinkhornChannelMixer for MI455X (gfx1250, wave32, WMMA bf16)
// Pipeline: pool -> fc1(silu) -> fc2 -> sinkhorn(LDS-resident) -> WMMA mix
// ---------------------------------------------------------------------------

#define C_DIM   256
#define HW_DIM  4096
#define B_DIM   32
#define R_DIM   64
#define KP      261        // padded Sinkhorn LDS row stride (gcd(261%64,64)=1)

typedef __attribute__((ext_vector_type(16))) __bf16 v16bf;
typedef __attribute__((ext_vector_type(2)))  __bf16 v2bf;
typedef __attribute__((ext_vector_type(8)))  float  v8f;

typedef union {
    v16bf v;
    uint4 u[2];
} v16bf_u;

__device__ __forceinline__ unsigned short f2bf(float f) {
    unsigned int u = __float_as_uint(f);
    unsigned int r = u + 0x7FFFu + ((u >> 16) & 1u);   // round-to-nearest-even
    return (unsigned short)(r >> 16);
}

// pack two f32 -> packed bf16x2 (v_cvt_pk_bf16_f32 when available)
__device__ __forceinline__ unsigned int pack_bf16(float lo, float hi) {
#if defined(__AMDGCN__) && __has_builtin(__builtin_amdgcn_cvt_pk_bf16_f32)
    union { v2bf v; unsigned int u; } cvt;
    cvt.v = __builtin_amdgcn_cvt_pk_bf16_f32(lo, hi);
    return cvt.u;
#else
    return (unsigned)f2bf(lo) | ((unsigned)f2bf(hi) << 16);
#endif
}

// ---------------------------------------------------------------------------
// Kernel 1: pooled[b,c] = mean over HW.  One block per (b,c) row.
// ---------------------------------------------------------------------------
__global__ __launch_bounds__(256) void pool_kernel(const float* __restrict__ x,
                                                   float* __restrict__ pooled) {
    int bc = blockIdx.x;                       // 0..8191
    const float* p = x + (size_t)bc * HW_DIM;
    int tid = threadIdx.x;
    float s = 0.0f;
    #pragma unroll
    for (int i = tid; i < HW_DIM / 4; i += 256) {
        float4 f = *(const float4*)(p + i * 4);
        s += f.x + f.y + f.z + f.w;
    }
    __shared__ float red[256];
    red[tid] = s;
    __syncthreads();
    for (int off = 128; off > 0; off >>= 1) {
        if (tid < off) red[tid] += red[tid + off];
        __syncthreads();
    }
    if (tid == 0) pooled[bc] = red[0] * (1.0f / (float)HW_DIM);
}

// ---------------------------------------------------------------------------
// Kernel 2: h[b,r] = silu(pooled[b,:] . w1[r,:] + b1[r])
// ---------------------------------------------------------------------------
__global__ __launch_bounds__(256) void fc1_kernel(const float* __restrict__ pooled,
                                                  const float* __restrict__ w1,
                                                  const float* __restrict__ b1,
                                                  float* __restrict__ h) {
    int idx = blockIdx.x * 256 + threadIdx.x;  // 0..2047
    if (idx >= B_DIM * R_DIM) return;
    int b = idx >> 6;
    int r = idx & 63;
    const float* pv = pooled + b * C_DIM;
    const float* wv = w1 + r * C_DIM;
    float acc = b1[r];
    #pragma unroll 4
    for (int i = 0; i < C_DIM; i += 4) {
        float4 a = *(const float4*)(pv + i);
        float4 w = *(const float4*)(wv + i);
        acc += a.x * w.x + a.y * w.y + a.z * w.z + a.w * w.w;
    }
    float sig = 1.0f / (1.0f + __expf(-acc));
    h[idx] = acc * sig;
}

// ---------------------------------------------------------------------------
// Kernel 3: weights[b, o] = h[b,:] . w2[o,:] + b2[o]   (o in [0, 65536))
// Each thread keeps one w2 row (64 f32) in registers, reuses it for all 32 b.
// ---------------------------------------------------------------------------
__global__ __launch_bounds__(256) void fc2_kernel(const float* __restrict__ h,
                                                  const float* __restrict__ w2,
                                                  const float* __restrict__ b2,
                                                  float* __restrict__ weights) {
    __shared__ float hs[B_DIM * R_DIM];
    int tid = threadIdx.x;
    for (int i = tid; i < B_DIM * R_DIM; i += 256) hs[i] = h[i];
    __syncthreads();

    int o = blockIdx.x * 256 + tid;            // 0..65535
    const float* wp = w2 + (size_t)o * R_DIM;
    float wrow[R_DIM];
    #pragma unroll
    for (int i = 0; i < R_DIM; i += 4) {
        float4 f = *(const float4*)(wp + i);
        wrow[i] = f.x; wrow[i + 1] = f.y; wrow[i + 2] = f.z; wrow[i + 3] = f.w;
    }
    float bias = b2[o];
    for (int b = 0; b < B_DIM; ++b) {
        const float* hv = hs + b * R_DIM;
        float acc = bias;
        #pragma unroll
        for (int i = 0; i < R_DIM; ++i) acc += hv[i] * wrow[i];
        weights[(size_t)b * (C_DIM * C_DIM) + o] = acc;
    }
}

// ---------------------------------------------------------------------------
// Kernel 4: Sinkhorn-Knopp, one 256x256 matrix per block, fully LDS-resident.
// Row stride padded to 261 floats -> conflict-free row/col/elementwise access
// (261 mod 64 = 5, gcd(5,64)=1).  Two-stage reductions use all 1024 threads.
// Emits bf16 M for the WMMA stage.
// ---------------------------------------------------------------------------
__global__ __launch_bounds__(1024) void sinkhorn_kernel(const float* __restrict__ weights,
                                                        unsigned short* __restrict__ Mbf) {
    extern __shared__ float sm[];
    float* Ksh  = sm;                          // 256 * KP floats (padded matrix)
    float* part = sm + C_DIM * KP;             // 1024 partials
    float* rsum = part + 1024;                 // 256
    float* csum = rsum + C_DIM;                // 256

    int b    = blockIdx.x;
    int tid  = threadIdx.x;
    int line = tid & 255;                      // row/col index for reductions
    int quar = tid >> 8;                       // 0..3 : 64-element quarter
    const float* W = weights + (size_t)b * (C_DIM * C_DIM);

    // load + nan_to_num (nan/±inf -> 0)
    for (int i = tid; i < C_DIM * C_DIM; i += 1024) {
        float v = W[i];
        unsigned int u = __float_as_uint(v);
        if ((u & 0x7f800000u) == 0x7f800000u) v = 0.0f;
        Ksh[(i >> 8) * KP + (i & 255)] = v;
    }
    __syncthreads();

    // row max (two-stage)
    {
        const float* rp = Ksh + line * KP + quar * 64;
        float m = rp[0];
        for (int j = 1; j < 64; ++j) m = fmaxf(m, rp[j]);
        part[quar * 256 + line] = m;
    }
    __syncthreads();
    if (tid < C_DIM)
        rsum[tid] = fmaxf(fmaxf(part[tid], part[tid + 256]),
                          fmaxf(part[tid + 512], part[tid + 768]));
    __syncthreads();

    // exp(v - rowmax)
    for (int i = tid; i < C_DIM * C_DIM; i += 1024) {
        int r = i >> 8, cc = i & 255;
        Ksh[r * KP + cc] = __expf(Ksh[r * KP + cc] - rsum[r]);
    }
    __syncthreads();

    // 20 iterations of row / column normalization
    for (int it = 0; it < 20; ++it) {
        {   // row sums
            const float* rp = Ksh + line * KP + quar * 64;
            float s = 0.0f;
            for (int j = 0; j < 64; ++j) s += rp[j];
            part[quar * 256 + line] = s;
        }
        __syncthreads();
        if (tid < C_DIM)
            rsum[tid] = 1.0f / (part[tid] + part[tid + 256] + part[tid + 512]
                                + part[tid + 768] + 1e-12f);
        __syncthreads();
        for (int i = tid; i < C_DIM * C_DIM; i += 1024) {
            int r = i >> 8, cc = i & 255;
            Ksh[r * KP + cc] *= rsum[r];
        }
        __syncthreads();
        {   // column sums
            const float* cp = Ksh + (quar * 64) * KP + line;
            float s = 0.0f;
            for (int j = 0; j < 64; ++j) s += cp[j * KP];
            part[quar * 256 + line] = s;
        }
        __syncthreads();
        if (tid < C_DIM)
            csum[tid] = 1.0f / (part[tid] + part[tid + 256] + part[tid + 512]
                                + part[tid + 768] + 1e-12f);
        __syncthreads();
        for (int i = tid; i < C_DIM * C_DIM; i += 1024) {
            int r = i >> 8, cc = i & 255;
            Ksh[r * KP + cc] *= csum[cc];
        }
        __syncthreads();
    }

    // emit bf16 (packed pairs)
    unsigned int* outp = (unsigned int*)(Mbf + (size_t)b * (C_DIM * C_DIM));
    for (int i = tid; i < (C_DIM * C_DIM) / 2; i += 1024) {
        int r = i >> 7, cc = (i & 127) * 2;
        outp[i] = pack_bf16(Ksh[r * KP + cc], Ksh[r * KP + cc + 1]);
    }
}

// ---------------------------------------------------------------------------
// Kernel 5: out = x + scale * (M @ x_flat), bf16 WMMA with f32 accumulation.
// Block tile: 32 (M) x 256 (N); 8 waves as a 2x4 grid, each wave owns a
// 16 (M) x 64 (N) strip = 4 WMMA tiles sharing one A fragment per K chunk.
// K looped in chunks of 32 (v_wmma_f32_16x16x32_bf16).
// ---------------------------------------------------------------------------
__global__ __launch_bounds__(256) void mix_kernel(const float* __restrict__ x,
                                                  const unsigned short* __restrict__ Mbf,
                                                  const float* __restrict__ scale,
                                                  float* __restrict__ out) {
    // Padded LDS tiles: rows 16B-aligned for ds_load_b128 fragment reads.
    __shared__ __align__(16) unsigned short Ash[32][40];    // 32 x 32 bf16 (+pad)
    __shared__ __align__(16) unsigned short Bsh[32][264];   // 32 x 256 bf16 (+pad)

    int b  = blockIdx.z;                 // batch
    int m0 = blockIdx.y * 32;            // M tile origin
    int n0 = blockIdx.x * 256;           // N tile origin
    int tid  = threadIdx.x;
    int wave = tid >> 5;                 // 0..7
    int lane = tid & 31;
    int mw   = (wave >> 2) * 16;         // 0 or 16 : wave M sub-origin
    int nw   = (wave & 3) * 64;          // 0,64,128,192 : wave N sub-origin

    const unsigned short* Mb = Mbf + (size_t)b * (C_DIM * C_DIM);
    const float* Xb = x + (size_t)b * C_DIM * HW_DIM;

    v8f acc[4] = {};

    for (int k0 = 0; k0 < C_DIM; k0 += 32) {
        // --- stage A: 32 rows x 32 K of bf16 Sinkhorn matrix (uint2/thread) ---
        {
            int r  = tid >> 3;           // 0..31
            int kk = (tid & 7) * 4;      // 0,4,..,28
            uint2 val = *(const uint2*)(Mb + (m0 + r) * C_DIM + k0 + kk);
            *(uint2*)&Ash[r][kk] = val;
        }
        // --- stage B: x rows k0..k0+31, cols n0..n0+255, f32 -> bf16 ---
        {
            int kk   = tid >> 3;               // 0..31
            int cseg = (tid & 7) * 32;         // 0,32,..,224
            const float* src = Xb + (size_t)(k0 + kk) * HW_DIM + n0 + cseg;
            #pragma unroll
            for (int j = 0; j < 8; ++j) {
                float4 f = *(const float4*)(src + j * 4);
                *(unsigned int*)&Bsh[kk][cseg + j * 4]     = pack_bf16(f.x, f.y);
                *(unsigned int*)&Bsh[kk][cseg + j * 4 + 2] = pack_bf16(f.z, f.w);
            }
        }
        __syncthreads();

        // --- A fragment (ISA 16-bit A 16x32 layout):
        // lanes 0-15 : row=lane,    K = e<8 ? e   : 16+(e-8)
        // lanes 16-31: row=lane-16, K = e<8 ? 8+e : 24+(e-8)
        int r     = mw + (lane & 15);
        int kbase = (lane & 16) ? 8 : 0;
        v16bf_u a;
        a.u[0] = *(const uint4*)&Ash[r][kbase];        // K = kbase .. kbase+7
        a.u[1] = *(const uint4*)&Ash[r][kbase + 16];   // K = kbase+16 .. kbase+23

        // --- 4 B fragments share this A fragment ---
        #pragma unroll
        for (int t = 0; t < 4; ++t) {
            v16bf_u bb;   // lane holds column N=lane&15 of sub-tile t, K in A's order
            int bn = nw + t * 16 + (lane & 15);
            #pragma unroll
            for (int s = 0; s < 16; ++s)
                ((unsigned short*)&bb)[s] = Bsh[kbase + (s & 7) + ((s >> 3) << 4)][bn];
            acc[t] = __builtin_amdgcn_wmma_f32_16x16x32_bf16(
                        /*neg_a=*/false, a.v, /*neg_b=*/false, bb.v,
                        /*c_mod=*/(short)0, acc[t], /*reuse_a=*/false, /*reuse_b=*/false);
        }
        __syncthreads();
    }

    // --- fused residual writeback: out = x + scale * acc ---
    // C/D layout: VGPR v -> M = v (lanes 0-15) or v+8 (lanes 16-31); N = lane&15
    // Residuals gathered into a temp array first so loads/stores form clauses.
    float sc = scale[0];
    int mb = m0 + mw + ((lane >> 4) << 3);
    #pragma unroll
    for (int t = 0; t < 4; ++t) {
        int n = n0 + nw + t * 16 + (lane & 15);
        const float* xp = Xb + (size_t)mb * HW_DIM + n;
        float*       op = out + ((size_t)b * C_DIM + mb) * HW_DIM + n;
        float res[8];
        #pragma unroll
        for (int v = 0; v < 8; ++v) res[v] = xp[(size_t)v * HW_DIM];
        #pragma unroll
        for (int v = 0; v < 8; ++v) op[(size_t)v * HW_DIM] = res[v] + sc * acc[t][v];
    }
}

// ---------------------------------------------------------------------------
// Launch
// ---------------------------------------------------------------------------
extern "C" void kernel_launch(void* const* d_in, const int* in_sizes, int n_in,
                              void* d_out, int out_size, void* d_ws, size_t ws_size,
                              hipStream_t stream) {
    const float* x     = (const float*)d_in[0];
    const float* w1    = (const float*)d_in[1];
    const float* b1    = (const float*)d_in[2];
    const float* w2    = (const float*)d_in[3];
    const float* b2    = (const float*)d_in[4];
    const float* scale = (const float*)d_in[5];
    float* out = (float*)d_out;

    char* ws = (char*)d_ws;
    float*          pooled  = (float*)(ws);                              // 32 KB
    float*          h       = (float*)(ws + 32768);                      // 8 KB
    float*          weights = (float*)(ws + 65536);                      // 8 MB
    unsigned short* Mbf     = (unsigned short*)(ws + 65536 + 8388608);   // 4 MB

    pool_kernel<<<B_DIM * C_DIM, 256, 0, stream>>>(x, pooled);
    fc1_kernel<<<(B_DIM * R_DIM + 255) / 256, 256, 0, stream>>>(pooled, w1, b1, h);
    fc2_kernel<<<(C_DIM * C_DIM) / 256, 256, 0, stream>>>(h, w2, b2, weights);

    size_t lds = (size_t)(C_DIM * KP + 1024 + 2 * C_DIM) * sizeof(float); // ~273 KB
    sinkhorn_kernel<<<B_DIM, 1024, lds, stream>>>(weights, Mbf);

    dim3 grid(HW_DIM / 256, C_DIM / 32, B_DIM);   // 16 x 8 x 32
    mix_kernel<<<grid, 256, 0, stream>>>(x, Mbf, scale, out);
}
